// LoRALinear_74139725463581
// MI455X (gfx1250) — matrix operand order repaired and checked by reading, not compiled, not run on hardware
//
#include <hip/hip_runtime.h>
#include <hip/hip_bf16.h>

// ---- problem constants (fixed by the reference) ----
#define T_TOK   8192
#define DIN     4096
#define DOUT    4096
#define NUM_AD  4
#define MAX_R   64
#define NPAGES  256   // NUM_AD * MAX_R

// ---- tiling ----
#define BM 128
#define BN 128          // kernels 2/3
#define BN1 256         // kernel 1 (base GEMM): 128x256 block, 64x64 wave tiles
#define BK 32
#define LDW 20          // uints per LDS tile row (80B stride -> conflict-free ds_read_b128)

typedef __attribute__((ext_vector_type(16))) __bf16 bf16x16;
typedef __attribute__((ext_vector_type(2)))  __bf16 bf16x2;
typedef __attribute__((ext_vector_type(8)))  float  f32x8;
typedef __attribute__((ext_vector_type(4)))  unsigned int u32x4;
typedef __attribute__((ext_vector_type(8)))  int    i32x8;
typedef __attribute__((ext_vector_type(4)))  int    i32x4;

struct B32x { uint4 lo, hi; };

// A-matrix fragment: per-lane two 16B chunks 32B apart (interleaved K layout, ISA 7.12.2)
__device__ __forceinline__ bf16x16 ld_frag_a(const unsigned* p) {
  B32x f; f.lo = *(const uint4*)p; f.hi = *(const uint4*)(p + 8);
  return __builtin_bit_cast(bf16x16, f);
}
// B-matrix fragment: per-lane two 16B chunks 32B apart (interleaved K layout, same as A)
__device__ __forceinline__ bf16x16 ld_frag_b(const unsigned* p) {
  B32x f; f.lo = *(const uint4*)p; f.hi = *(const uint4*)(p + 8);
  return __builtin_bit_cast(bf16x16, f);
}
// packed f32x2 -> bf16x2 (lowers to v_cvt_pk_bf16_f32)
__device__ __forceinline__ unsigned pk2(float a, float b) {
  bf16x2 v; v.x = (__bf16)a; v.y = (__bf16)b;
  return __builtin_bit_cast(unsigned, v);
}
__device__ __forceinline__ unsigned short bf1(float a) {
  return __builtin_bit_cast(unsigned short, (__bf16)a);
}

// ---------------------------------------------------------------------
// Tensor Data Mover: 2D bf16 tile (tile_d1 rows x tile_d0 elems) from
// global to LDS, padding 16B after each 64B row => LDS row stride 80B.
// ---------------------------------------------------------------------
#if defined(__HIP_DEVICE_COMPILE__) && __has_builtin(__builtin_amdgcn_tensor_load_to_lds)
#define HAVE_TDM 1
__device__ __forceinline__ void tdm_load_tile_bf16(unsigned lds_addr, const void* gaddr,
                                                   unsigned tile_d0, unsigned tile_d1,
                                                   unsigned stride_d0_elems) {
  unsigned long long ga = (unsigned long long)(size_t)gaddr;
  u32x4 g0;
  g0.x = 1u;                                          // count=1 (valid user D#)
  g0.y = lds_addr;                                    // LDS byte address
  g0.z = (unsigned)(ga & 0xFFFFFFFFu);                // global_addr[31:0]
  g0.w = (unsigned)((ga >> 32) & 0x01FFFFFFu)         // global_addr[56:32]
       | (2u << 30);                                  // type = 2 ("image")
  i32x8 g1;
  g1[0] = (int)((1u << 16)                            // data_size = 2 bytes
              | (1u << 20)                            // pad_enable
              | (3u << 22)                            // pad_interval: 16 DWORDs (64B)
              | (3u << 25));                          // pad_amount:   4 DWORDs (16B)
  g1[1] = (int)((tile_d0 & 0xFFFFu) << 16);           // tensor_dim0 = tile_d0 (no OOB)
  g1[2] = (int)(((tile_d1 & 0xFFFFu) << 16));         // tensor_dim1 = tile_d1
  g1[3] = (int)(tile_d0 << 16);                       // tile_dim0
  g1[4] = (int)(tile_d1 & 0xFFFFu);                   // tile_dim1 (tile_dim2 = 0)
  g1[5] = (int)stride_d0_elems;                       // tensor_dim0_stride[31:0]
  g1[6] = 0;                                          // stride hi / dim1_stride lo
  g1[7] = 0;
  i32x4 z4 = {0, 0, 0, 0};
#if __clang_major__ >= 23
  i32x8 z8 = {0, 0, 0, 0, 0, 0, 0, 0};
  __builtin_amdgcn_tensor_load_to_lds(g0, g1, z4, z4, z8, 0);
#else
  __builtin_amdgcn_tensor_load_to_lds(g0, g1, z4, z4, 0);
#endif
  __builtin_amdgcn_s_wait_tensorcnt(0);
}
#endif

// =====================================================================
// Kernel 1: base = x @ W^T + bias     (M=8192, N=4096, K=4096)
// 128x256 block, 8 waves, each wave computes 64x64 (4x4 fragments,
// 16 WMMA per k-step from 8 fragment loads => 1.0 ds_read/WMMA)
// =====================================================================
__global__ __launch_bounds__(256)
void base_gemm_kernel(const float* __restrict__ x, const float* __restrict__ w,
                      const float* __restrict__ bias, float* __restrict__ out) {
  __shared__ unsigned sx[BM  * LDW];                  // 128 token rows
  __shared__ unsigned sw[BN1 * LDW];                  // 256 output-col rows
  const int tid  = threadIdx.x;
  const int lane = tid & 31, wv = tid >> 5;
  const int wm = wv & 1, wn = wv >> 1;                // 2 (M) x 4 (N) waves, 64x64 each
  const int lm = lane & 15, sel = lane >> 4;
  const int row0 = blockIdx.y * BM;
  const int n0   = blockIdx.x * BN1;

  f32x8 acc[4][4] = {};

  for (int k0 = 0; k0 < DIN; k0 += BK) {
    // stage x tile: 128x32 f32 -> bf16 (4 float4 chunks per thread)
    #pragma unroll
    for (int i = 0; i < 4; ++i) {
      int cid = tid + i * 256;
      int r = cid >> 3, c4 = cid & 7;
      float4 vx = *(const float4*)(x + (long)(row0 + r) * DIN + k0 + c4 * 4);
      uint2 tx; tx.x = pk2(vx.x, vx.y); tx.y = pk2(vx.z, vx.w);
      *(uint2*)&sx[r * LDW + c4 * 2] = tx;
      if (k0 + BK < DIN)
        __builtin_prefetch(x + (long)(row0 + r) * DIN + k0 + BK + c4 * 4, 0, 3);
    }
    // stage W tile: 256x32 f32 -> bf16 (8 float4 chunks per thread)
    #pragma unroll
    for (int i = 0; i < 8; ++i) {
      int cid = tid + i * 256;
      int r = cid >> 3, c4 = cid & 7;
      float4 vw = *(const float4*)(w + (long)(n0 + r) * DIN + k0 + c4 * 4);
      uint2 tw; tw.x = pk2(vw.x, vw.y); tw.y = pk2(vw.z, vw.w);
      *(uint2*)&sw[r * LDW + c4 * 2] = tw;
      if (k0 + BK < DIN)
        __builtin_prefetch(w + (long)(n0 + r) * DIN + k0 + BK + c4 * 4, 0, 3);
    }
    __syncthreads();
    bf16x16 af[4], bfr[4];
    #pragma unroll
    for (int mi = 0; mi < 4; ++mi)
      af[mi] = ld_frag_a(&sx[(wm * 64 + mi * 16 + lm) * LDW + sel * 4]);
    #pragma unroll
    for (int ni = 0; ni < 4; ++ni)
      bfr[ni] = ld_frag_b(&sw[(wn * 64 + ni * 16 + lm) * LDW + sel * 4]);
    #pragma unroll
    for (int mi = 0; mi < 4; ++mi)
      #pragma unroll
      for (int ni = 0; ni < 4; ++ni)
        acc[mi][ni] = __builtin_amdgcn_wmma_f32_16x16x32_bf16(
            false, af[mi], false, bfr[ni], (short)0, acc[mi][ni], false, false);
    __syncthreads();
  }
  #pragma unroll
  for (int ni = 0; ni < 4; ++ni) {
    int col = n0 + wn * 64 + ni * 16 + lm;
    float bv = bias[col];
    #pragma unroll
    for (int mi = 0; mi < 4; ++mi) {
      int rb = row0 + wm * 64 + mi * 16 + sel * 8;
      #pragma unroll
      for (int j = 0; j < 8; ++j)
        out[(long)(rb + j) * DOUT + col] = acc[mi][ni][j] + bv;
    }
  }
}

// =====================================================================
// Kernel 2: XA[t, a*64+r] = mask(t,a,r) * (x[t] . a_cache[page[a,r]])
//           (M=8192, N=256, K=4096) -> bf16 workspace
// =====================================================================
__global__ __launch_bounds__(256)
void xa_gemm_kernel(const float* __restrict__ x, const float* __restrict__ a_cache,
                    const int* __restrict__ page_tbl, const int* __restrict__ adapter_ids,
                    const int* __restrict__ ranks, unsigned short* __restrict__ xa) {
  __shared__ unsigned sx[BM * LDW];
  __shared__ unsigned sw[BN * LDW];
  __shared__ int s_pg[BN];
  __shared__ int s_aid[BM];
  __shared__ int s_rank[NUM_AD];
  const int tid  = threadIdx.x;
  const int lane = tid & 31, wv = tid >> 5;
  const int wm = wv >> 2, wn = wv & 3;
  const int lm = lane & 15, sel = lane >> 4;
  const int row0 = blockIdx.y * BM;
  const int n0   = blockIdx.x * BN;     // 0 or 128

  if (tid < BN)     s_pg[tid]  = page_tbl[n0 + tid];
  if (tid < BM)     s_aid[tid] = adapter_ids[row0 + tid];
  if (tid < NUM_AD) s_rank[tid] = ranks[tid];
  __syncthreads();

  f32x8 acc[4][2] = {};

  for (int k0 = 0; k0 < DIN; k0 += BK) {
    #pragma unroll
    for (int i = 0; i < 4; ++i) {
      int cid = tid + i * 256;
      int r = cid >> 3, c4 = cid & 7;
      float4 vx = *(const float4*)(x + (long)(row0 + r) * DIN + k0 + c4 * 4);
      float4 vw = *(const float4*)(a_cache + (long)s_pg[r] * DIN + k0 + c4 * 4);
      uint2 tx; tx.x = pk2(vx.x, vx.y); tx.y = pk2(vx.z, vx.w);
      uint2 tw; tw.x = pk2(vw.x, vw.y); tw.y = pk2(vw.z, vw.w);
      *(uint2*)&sx[r * LDW + c4 * 2] = tx;
      *(uint2*)&sw[r * LDW + c4 * 2] = tw;
    }
    __syncthreads();
    bf16x16 af[4], bfr[2];
    #pragma unroll
    for (int mi = 0; mi < 4; ++mi)
      af[mi] = ld_frag_a(&sx[(wm * 64 + mi * 16 + lm) * LDW + sel * 4]);
    #pragma unroll
    for (int ni = 0; ni < 2; ++ni)
      bfr[ni] = ld_frag_b(&sw[(wn * 32 + ni * 16 + lm) * LDW + sel * 4]);
    #pragma unroll
    for (int mi = 0; mi < 4; ++mi)
      #pragma unroll
      for (int ni = 0; ni < 2; ++ni)
        acc[mi][ni] = __builtin_amdgcn_wmma_f32_16x16x32_bf16(
            false, af[mi], false, bfr[ni], (short)0, acc[mi][ni], false, false);
    __syncthreads();
  }
  // masked bf16 epilogue: zero where adapter doesn't match or r >= rank
  #pragma unroll
  for (int ni = 0; ni < 2; ++ni) {
    int col = n0 + wn * 32 + ni * 16 + lm;
    int aC = col >> 6, rr = col & 63;
    int rank_ok = (rr < s_rank[aC]);
    #pragma unroll
    for (int mi = 0; mi < 4; ++mi) {
      int relR = wm * 64 + mi * 16 + sel * 8;
      #pragma unroll
      for (int j = 0; j < 8; ++j) {
        int keep = rank_ok && (s_aid[relR + j] == aC);
        xa[(long)(row0 + relR + j) * NPAGES + col] = keep ? bf1(acc[mi][ni][j])
                                                          : (unsigned short)0;
      }
    }
  }
}

// =====================================================================
// Kernel 3: out += XA @ B_gathered   (M=8192, N=4096, K=256)
//           A tile comes in via the Tensor Data Mover (bf16, padded rows);
//           B[k][n] = b_cache[page[k]][n]  (gather + transpose into LDS)
// =====================================================================
__global__ __launch_bounds__(256)
void lora_b_kernel(const unsigned short* __restrict__ xa, const float* __restrict__ b_cache,
                   const int* __restrict__ page_tbl, float* __restrict__ out) {
  __shared__ unsigned sx[BM * LDW];
  __shared__ unsigned sw[BN * LDW];
  __shared__ int s_pg[NPAGES];
  const int tid  = threadIdx.x;
  const int lane = tid & 31, wv = tid >> 5;
  const int wm = wv >> 2, wn = wv & 3;
  const int lm = lane & 15, sel = lane >> 4;
  const int row0 = blockIdx.y * BM;
  const int n0   = blockIdx.x * BN;

  if (tid < NPAGES) s_pg[tid] = page_tbl[tid];
#if defined(HAVE_TDM)
  const unsigned sx_lds = (unsigned)(size_t)(void*)&sx[0];   // LDS byte offset (low 32b)
#endif
  __syncthreads();

  f32x8 acc[4][2] = {};

  for (int k0 = 0; k0 < NPAGES; k0 += BK) {       // 8 k-steps
#if defined(HAVE_TDM)
    // A tile: TDM DMA, 128 rows x 32 bf16, 64B data + 16B pad per row (stride 80B)
    if (wv == 0)
      tdm_load_tile_bf16(sx_lds, xa + (long)row0 * NPAGES + k0,
                         /*tile_d0=*/BK, /*tile_d1=*/BM, /*stride=*/NPAGES);
#else
    #pragma unroll
    for (int i = 0; i < 2; ++i) {
      int cid = tid + i * 256;
      int r = cid >> 2, c = cid & 3;
      uint4 v = *(const uint4*)(xa + (long)(row0 + r) * NPAGES + k0 + c * 8);
      *(uint4*)&sx[r * LDW + c * 4] = v;
    }
#endif
    // B tile: 32(k) x 128(n) f32, gathered by page, stored transposed [n][k] bf16
    unsigned short* swh = (unsigned short*)sw;
    #pragma unroll
    for (int i = 0; i < 4; ++i) {
      int cid = tid + i * 256;
      int r = cid >> 5, c4 = cid & 31;            // r = k within tile, c4 = n chunk
      int pg = s_pg[k0 + r];
      float4 v = *(const float4*)(b_cache + (long)pg * DOUT + n0 + c4 * 4);
      int nb = c4 * 4;
      swh[(nb + 0) * (2 * LDW) + r] = bf1(v.x);
      swh[(nb + 1) * (2 * LDW) + r] = bf1(v.y);
      swh[(nb + 2) * (2 * LDW) + r] = bf1(v.z);
      swh[(nb + 3) * (2 * LDW) + r] = bf1(v.w);
    }
    __syncthreads();
    bf16x16 af[4], bfr[2];
    #pragma unroll
    for (int mi = 0; mi < 4; ++mi)
      af[mi] = ld_frag_a(&sx[(wm * 64 + mi * 16 + lm) * LDW + sel * 4]);
    #pragma unroll
    for (int ni = 0; ni < 2; ++ni)
      bfr[ni] = ld_frag_b(&sw[(wn * 32 + ni * 16 + lm) * LDW + sel * 4]);
    #pragma unroll
    for (int mi = 0; mi < 4; ++mi)
      #pragma unroll
      for (int ni = 0; ni < 2; ++ni)
        acc[mi][ni] = __builtin_amdgcn_wmma_f32_16x16x32_bf16(
            false, af[mi], false, bfr[ni], (short)0, acc[mi][ni], false, false);
    __syncthreads();
  }
  #pragma unroll
  for (int ni = 0; ni < 2; ++ni) {
    int col = n0 + wn * 32 + ni * 16 + lm;
    #pragma unroll
    for (int mi = 0; mi < 4; ++mi) {
      int rb = row0 + wm * 64 + mi * 16 + sel * 8;
      #pragma unroll
      for (int j = 0; j < 8; ++j) {
        long idx = (long)(rb + j) * DOUT + col;
        out[idx] = out[idx] + acc[mi][ni][j];
      }
    }
  }
}

// =====================================================================
extern "C" void kernel_launch(void* const* d_in, const int* in_sizes, int n_in,
                              void* d_out, int out_size, void* d_ws, size_t ws_size,
                              hipStream_t stream) {
  const float* x        = (const float*)d_in[0];
  const float* weight   = (const float*)d_in[1];
  const float* bias     = (const float*)d_in[2];
  const float* a_cache  = (const float*)d_in[3];
  const float* b_cache  = (const float*)d_in[4];
  const int* adapter_ids = (const int*)d_in[5];
  const int* page_tbl    = (const int*)d_in[6];
  const int* ranks       = (const int*)d_in[7];
  float* out = (float*)d_out;
  unsigned short* xa = (unsigned short*)d_ws;   // 8192 * 256 bf16 = 4 MB scratch

  dim3 blk(256);
  base_gemm_kernel<<<dim3(DOUT / BN1, T_TOK / BM), blk, 0, stream>>>(x, weight, bias, out);
  xa_gemm_kernel<<<dim3(NPAGES / BN, T_TOK / BM), blk, 0, stream>>>(
      x, a_cache, page_tbl, adapter_ids, ranks, xa);
  lora_b_kernel<<<dim3(DOUT / BN, T_TOK / BM), blk, 0, stream>>>(xa, b_cache, page_tbl, out);
}
